// MultiHeadDotProductAttention_19610820673940
// MI455X (gfx1250) — matrix operand order repaired and checked by reading, not compiled, not run on hardware
//
#include <hip/hip_runtime.h>

// ---------------------------------------------------------------------------
// T5-style causal MHA for MI455X (gfx1250), bf16 WMMA path.
//   B=4, S=2048, E=1024, H=16, D=64, NUM_BUCKETS=32, MAX_DISTANCE=128
// Pipeline:
//   1) qkv_gemm_kernel : Q/K/V = X @ W{q,k,v}  (f32 in -> bf16 [B,H,S,D] out)
//   2) flash_attn_kernel: fused QK^T + T5 bias + causal mask + online softmax
//                         + PV, bf16 WMMA, fp32 accum -> bf16 [B,S,H*D]
//   3) out_gemm_kernel  : out = attn @ Wo (bf16 x bf16 -> f32 [B,S,E])
// GEMM tiling: 128x128 block, 8 waves of 32x64 (2x4 fragments, 8 WMMA/K-step).
// bf16 tiles are staged with GLOBAL_LOAD_ASYNC_TO_LDS_B128 (ASYNCcnt);
// transposed stagings use packed dword LDS stores.
// ---------------------------------------------------------------------------

typedef __bf16 bf16;
typedef __attribute__((ext_vector_type(16))) __bf16 v16bf;
typedef __attribute__((ext_vector_type(8)))  float  v8f;

#define S_LEN 2048
#define EMB   1024
#define NH    16
#define HD    64
#define QT    64     // q rows per flash block
#define KT    32     // kv rows per flash tile
#define BSP   40     // padded n-major stride (80 B = 16B-aligned, banks spread)
#define TM    128    // GEMM block tile M
#define TN    128    // GEMM block tile N
#define TK    32     // GEMM K step

__device__ __forceinline__ v8f wmma_bf16(v16bf a, v16bf b, v8f c) {
  return __builtin_amdgcn_wmma_f32_16x16x32_bf16(false, a, false, b, (short)0, c,
                                                 false, false);
}

// Async 16B global->LDS copy (gfx1250 GLOBAL_LOAD_ASYNC_TO_LDS_B128, ASYNCcnt).
// Low 32 bits of a generic pointer to LDS are the LDS byte address (ISA 10.2).
__device__ __forceinline__ void async_copy16(void* lds_dst, const void* gsrc) {
  unsigned l = (unsigned)(uintptr_t)lds_dst;
  unsigned long long g = (unsigned long long)(uintptr_t)gsrc;
  asm volatile("global_load_async_to_lds_b128 %0, %1, off" ::"v"(l), "v"(g)
               : "memory");
}
__device__ __forceinline__ void async_wait0() {
  asm volatile("s_wait_asynccnt 0x0" ::: "memory");
}

// Convert two floats to bf16 and store as one dword (fuses to v_cvt_pk_bf16_f32).
__device__ __forceinline__ void store_bf16_pair(bf16* p, float a, float b) {
  unsigned short ua = __builtin_bit_cast(unsigned short, (bf16)a);
  unsigned short ub = __builtin_bit_cast(unsigned short, (bf16)b);
  *(unsigned*)p = (unsigned)ua | ((unsigned)ub << 16);
}

// A-matrix 16x32 fragment from row-major LDS tile.
// Layout used here (not the ISA table order): lane m=lane&15 is row; lane half kh selects K sub-block;
// VGPR v -> K = kh*16 + 2v (contiguous per lane half, matches load_b_frag).
__device__ __forceinline__ v16bf load_a_frag(const bf16* lds, int row0, int stride,
                                             int kbase, int lane) {
  int m = lane & 15, kh = lane >> 4;
  const bf16* rp = lds + (row0 + m) * stride + kbase;
  v16bf a;
#pragma unroll
  for (int v = 0; v < 8; ++v) {
    int k0 = kh * 16 + 2 * v;
    a[2 * v]     = rp[k0];
    a[2 * v + 1] = rp[k0 + 1];
  }
  return a;
}

// B-matrix 32x16 fragment. LDS tile stored "n-major": element (k,n) lives at
// lds[(col0+n)*stride + kbase + k]; each lane's half is 16 contiguous bf16,
// so with 16B-aligned strides this folds into two ds_load_b128.
__device__ __forceinline__ v16bf load_b_frag(const bf16* lds, int col0, int stride,
                                             int kbase, int lane) {
  int n = lane & 15, kh = lane >> 4;
  const bf16* cp = lds + (col0 + n) * stride + kbase;
  v16bf b;
#pragma unroll
  for (int v = 0; v < 8; ++v) {
    int k0 = kh * 16 + 2 * v;
    b[2 * v]     = cp[k0];
    b[2 * v + 1] = cp[k0 + 1];
  }
  return b;
}

// T5 causal bucket bias (NUM_BUCKETS=32, MAX_DISTANCE=128, max_exact=16).
__device__ __forceinline__ float t5_bias(const float* bias_s, int dist) {
  int bucket;
  if (dist < 16) {
    bucket = dist < 0 ? 0 : dist;
  } else {
    // 16 + log(d/16)/log(128/16) * 16 ; log(8) = 2.0794415
    float ls = __logf((float)dist * (1.0f / 16.0f)) * (16.0f / 2.0794415416798357f);
    bucket = 16 + (int)ls;
    if (bucket > 31) bucket = 31;
  }
  return bias_s[bucket];
}

// Stage W[k0..k0+31][n0..n0+127] f32 -> n-major bf16 LDS tile, paired K-rows.
// Thread owns 2 consecutive k rows x 8 n columns; 8 dword LDS stores.
__device__ __forceinline__ void stage_b_tile(bf16 (*Bs)[BSP], const float* W,
                                             int k0, int n0, int tid,
                                             bool prefetch_next) {
  int kk = (tid >> 4) * 2, nb = (tid & 15) * 8;
  const float* s0 = W + (size_t)(k0 + kk) * EMB + n0 + nb;
  const float* s1 = s0 + EMB;
  if (prefetch_next) __builtin_prefetch(s0 + (size_t)TK * EMB, 0, 1);
#pragma unroll
  for (int i = 0; i < 8; i += 4) {
    float4 f0 = *(const float4*)(s0 + i);
    float4 f1 = *(const float4*)(s1 + i);
    store_bf16_pair(&Bs[nb + i + 0][kk], f0.x, f1.x);
    store_bf16_pair(&Bs[nb + i + 1][kk], f0.y, f1.y);
    store_bf16_pair(&Bs[nb + i + 2][kk], f0.z, f1.z);
    store_bf16_pair(&Bs[nb + i + 3][kk], f0.w, f1.w);
  }
}

// ---------------------------------------------------------------------------
// Kernel 1: fused Q/K/V projection GEMMs. blockIdx.z selects which weight.
// C[8192 x 1024] tile per block: 128x128, 8 waves of 32x64.
// ---------------------------------------------------------------------------
__global__ __launch_bounds__(256) void qkv_gemm_kernel(
    const float* __restrict__ Xq, const float* __restrict__ Xkv,
    const float* __restrict__ Wq, const float* __restrict__ Wk,
    const float* __restrict__ Wv,
    bf16* __restrict__ q_ws, bf16* __restrict__ k_ws, bf16* __restrict__ v_ws) {
  __shared__ alignas(16) bf16 As[TM][TK];   // row-major activations (8 KB)
  __shared__ alignas(16) bf16 Bs[TN][BSP];  // n-major weights (10 KB)

  const int which = blockIdx.z;
  const float* X = (which == 0) ? Xq : Xkv;
  const float* W = (which == 0) ? Wq : (which == 1) ? Wk : Wv;
  bf16* dst = (which == 0) ? q_ws : (which == 1) ? k_ws : v_ws;

  const int m0 = blockIdx.x * TM;
  const int n0 = blockIdx.y * TN;
  const int tid = threadIdx.x;
  const int lane = tid & 31, w = tid >> 5;
  const int wm = (w >> 1) * 32, wn = (w & 1) * 64;

  v8f acc[2][4] = {};

  for (int k0 = 0; k0 < EMB; k0 += TK) {
    // Stage A: 128x32 f32 -> bf16 row-major; 16 elements/thread, paired stores.
    {
      int row = tid >> 1, kc = (tid & 1) * 16;
      const float* src = X + (size_t)(m0 + row) * EMB + k0 + kc;
      if (k0 + TK < EMB) __builtin_prefetch(src + TK, 0, 1);
#pragma unroll
      for (int i = 0; i < 16; i += 4) {
        float4 f = *(const float4*)(src + i);
        store_bf16_pair(&As[row][kc + i + 0], f.x, f.y);
        store_bf16_pair(&As[row][kc + i + 2], f.z, f.w);
      }
    }
    // Stage B transposed: W -> Bs[n][k], paired K-rows, dword stores.
    stage_b_tile(Bs, W, k0, n0, tid, k0 + TK < EMB);
    __syncthreads();

    v16bf a0 = load_a_frag(&As[0][0], wm, TK, 0, lane);
    v16bf a1 = load_a_frag(&As[0][0], wm + 16, TK, 0, lane);
#pragma unroll
    for (int j = 0; j < 4; ++j) {
      v16bf bj = load_b_frag(&Bs[0][0], wn + j * 16, BSP, 0, lane);
      acc[0][j] = wmma_bf16(a0, bj, acc[0][j]);
      acc[1][j] = wmma_bf16(a1, bj, acc[1][j]);
    }
    __syncthreads();
  }

  // Scatter bf16 results to [B, H, S, D] for the attention stage.
  const int half = lane >> 4, nn = lane & 15;
#pragma unroll
  for (int i = 0; i < 2; ++i)
#pragma unroll
    for (int j = 0; j < 4; ++j)
#pragma unroll
      for (int r = 0; r < 8; ++r) {
        int m = m0 + wm + i * 16 + half * 8 + r;  // b*S + s
        int n = n0 + wn + j * 16 + nn;            // h*64 + d
        int b = m >> 11, s = m & (S_LEN - 1);
        int h = n >> 6, d = n & (HD - 1);
        dst[((size_t)(b * NH + h) * S_LEN + s) * HD + d] = (bf16)acc[i][j][r];
      }
}

// ---------------------------------------------------------------------------
// Kernel 2: fused flash attention. Block = 4 waves, 64 q rows; KV tiles of 32.
// ---------------------------------------------------------------------------
__global__ __launch_bounds__(128) void flash_attn_kernel(
    const bf16* __restrict__ q_ws, const bf16* __restrict__ k_ws,
    const bf16* __restrict__ v_ws, const float* __restrict__ rel_bias,
    bf16* __restrict__ attn_ws) {
  __shared__ alignas(16) bf16 Qs[QT][HD];    // 8 KB
  __shared__ alignas(16) bf16 Ks[KT][HD];    // 4 KB (row-major == n-major)
  __shared__ alignas(16) bf16 Vs[HD][BSP];   // transposed V tile (5 KB)
  __shared__ alignas(16) bf16 Ps[4][16][KT]; // per-wave softmax P staging
  __shared__ float bias_s[32];

  const int bh = blockIdx.y;
  const int b = bh >> 4, h = bh & (NH - 1);
  const int qbase = blockIdx.x * QT;
  const int tid = threadIdx.x, lane = tid & 31, w = tid >> 5;
  const int half = lane >> 4, nn = lane & 15;

  const bf16* qp = q_ws + (size_t)bh * S_LEN * HD;
  const bf16* kp = k_ws + (size_t)bh * S_LEN * HD;
  const bf16* vp = v_ws + (size_t)bh * S_LEN * HD;

  if (tid < 32) bias_s[tid] = rel_bias[h * 32 + tid];

  // Stage this block's Q tile (64x64 bf16) via async global->LDS copies.
  {
    int row = tid >> 1, dc = (tid & 1) * 32;
    const bf16* src = qp + (size_t)(qbase + row) * HD + dc;
#pragma unroll
    for (int i = 0; i < 4; ++i)
      async_copy16(&Qs[row][dc + i * 8], src + i * 8);
  }
  async_wait0();
  __syncthreads();
  v16bf qf0 = load_a_frag(&Qs[0][0], w * 16, HD, 0, lane);
  v16bf qf1 = load_a_frag(&Qs[0][0], w * 16, HD, 32, lane);

  float m_run[8], l_run[8];
  v8f o[4] = {};
#pragma unroll
  for (int r = 0; r < 8; ++r) { m_run[r] = -1e30f; l_run[r] = 0.0f; }

  const int kv_end = qbase + QT;  // causal: keys beyond the q tile are masked
  for (int kv = 0; kv < kv_end; kv += KT) {
    __syncthreads();
    // Stage K (async, row-major) and V (transposed, paired dword stores).
    {
      int j = tid >> 2, doff = (tid & 3) * 16;
      const bf16* ksrc = kp + (size_t)(kv + j) * HD + doff;
      if (kv + KT < kv_end) __builtin_prefetch(ksrc + (size_t)KT * HD, 0, 1);
      async_copy16(&Ks[j][doff], ksrc);
      async_copy16(&Ks[j][doff + 8], ksrc + 8);
    }
    {
      int jp = tid & 15, dg = (tid >> 4) * 8;   // j pair, 8 d-rows
      int j0 = jp * 2;
      const bf16* v0 = vp + (size_t)(kv + j0) * HD + dg;
      const bf16* v1 = v0 + HD;
      if (kv + KT < kv_end) __builtin_prefetch(v0 + (size_t)KT * HD, 0, 1);
      union { uint4 v; unsigned u[4]; } r0, r1;
      r0.v = *(const uint4*)v0;
      r1.v = *(const uint4*)v1;
#pragma unroll
      for (int i = 0; i < 8; ++i) {
        unsigned lo = (r0.u[i >> 1] >> ((i & 1) * 16)) & 0xffffu;
        unsigned hi = (r1.u[i >> 1] >> ((i & 1) * 16)) & 0xffffu;
        *(unsigned*)&Vs[dg + i][j0] = lo | (hi << 16);
      }
    }
    async_wait0();
    __syncthreads();

    // S = (Q K^T) for two 16-col fragments, contracting D=64 in two chunks.
    v8f s0 = {}, s1 = {};
    {
      v16bf bk;
      bk = load_b_frag(&Ks[0][0], 0, HD, 0, lane);   s0 = wmma_bf16(qf0, bk, s0);
      bk = load_b_frag(&Ks[0][0], 0, HD, 32, lane);  s0 = wmma_bf16(qf1, bk, s0);
      bk = load_b_frag(&Ks[0][0], 16, HD, 0, lane);  s1 = wmma_bf16(qf0, bk, s1);
      bk = load_b_frag(&Ks[0][0], 16, HD, 32, lane); s1 = wmma_bf16(qf1, bk, s1);
    }

    // Scale + T5 bias + causal mask + online softmax (rowwise across 16 lanes).
#pragma unroll
    for (int r = 0; r < 8; ++r) {
      int row = qbase + w * 16 + half * 8 + r;
      int c0 = kv + nn, c1 = kv + 16 + nn;
      float v0 = s0[r] * 0.125f + t5_bias(bias_s, row - c0);
      float v1 = s1[r] * 0.125f + t5_bias(bias_s, row - c1);
      if (c0 > row) v0 = -1e30f;
      if (c1 > row) v1 = -1e30f;
      float rmax = fmaxf(v0, v1);
#pragma unroll
      for (int msk = 1; msk < 16; msk <<= 1)
        rmax = fmaxf(rmax, __shfl_xor(rmax, msk, 32));
      float mnew = fmaxf(m_run[r], rmax);
      float alpha = __expf(m_run[r] - mnew);
      float p0 = __expf(v0 - mnew);
      float p1 = __expf(v1 - mnew);
      float rsum = p0 + p1;
#pragma unroll
      for (int msk = 1; msk < 16; msk <<= 1) rsum += __shfl_xor(rsum, msk, 32);
      m_run[r] = mnew;
      l_run[r] = l_run[r] * alpha + rsum;
#pragma unroll
      for (int df = 0; df < 4; ++df) o[df][r] *= alpha;
      Ps[w][half * 8 + r][nn] = (bf16)p0;
      Ps[w][half * 8 + r][16 + nn] = (bf16)p1;
    }

    // O += P @ V  (A = P 16x32, B = V^T chunks 32x16, fp32 accumulate)
    v16bf pf = load_a_frag(&Ps[w][0][0], 0, KT, 0, lane);
#pragma unroll
    for (int df = 0; df < 4; ++df) {
      v16bf vb = load_b_frag(&Vs[0][0], df * 16, BSP, 0, lane);
      o[df] = wmma_bf16(pf, vb, o[df]);
    }
  }

  // Normalize and scatter to [B, S, H*D] (bf16) for the output projection.
#pragma unroll
  for (int r = 0; r < 8; ++r) {
    int row = qbase + w * 16 + half * 8 + r;
    float inv = 1.0f / l_run[r];
#pragma unroll
    for (int df = 0; df < 4; ++df)
      attn_ws[(size_t)(b * S_LEN + row) * EMB + h * HD + df * 16 + nn] =
          (bf16)(o[df][r] * inv);
  }
}

// ---------------------------------------------------------------------------
// Kernel 3: output projection, out = attn(bf16) @ Wo(f32->bf16) -> f32.
// ---------------------------------------------------------------------------
__global__ __launch_bounds__(256) void out_gemm_kernel(
    const bf16* __restrict__ A, const float* __restrict__ Wo,
    float* __restrict__ out) {
  __shared__ alignas(16) bf16 As[TM][TK];
  __shared__ alignas(16) bf16 Bs[TN][BSP];

  const int m0 = blockIdx.x * TM;
  const int n0 = blockIdx.y * TN;
  const int tid = threadIdx.x;
  const int lane = tid & 31, w = tid >> 5;
  const int wm = (w >> 1) * 32, wn = (w & 1) * 64;

  v8f acc[2][4] = {};

  for (int k0 = 0; k0 < EMB; k0 += TK) {
    // Stage A (bf16) via async global->LDS copies: 32 B per thread.
    {
      int row = tid >> 1, kc = (tid & 1) * 16;
      const bf16* src = A + (size_t)(m0 + row) * EMB + k0 + kc;
      if (k0 + TK < EMB) __builtin_prefetch(src + TK, 0, 1);
      async_copy16(&As[row][kc], src);
      async_copy16(&As[row][kc + 8], src + 8);
    }
    // Stage B transposed: Wo -> Bs[n][k], paired K-rows, dword stores.
    stage_b_tile(Bs, Wo, k0, n0, tid, k0 + TK < EMB);
    async_wait0();
    __syncthreads();

    v16bf a0 = load_a_frag(&As[0][0], wm, TK, 0, lane);
    v16bf a1 = load_a_frag(&As[0][0], wm + 16, TK, 0, lane);
#pragma unroll
    for (int j = 0; j < 4; ++j) {
      v16bf bj = load_b_frag(&Bs[0][0], wn + j * 16, BSP, 0, lane);
      acc[0][j] = wmma_bf16(a0, bj, acc[0][j]);
      acc[1][j] = wmma_bf16(a1, bj, acc[1][j]);
    }
    __syncthreads();
  }

  const int half = lane >> 4, nn = lane & 15;
#pragma unroll
  for (int i = 0; i < 2; ++i)
#pragma unroll
    for (int j = 0; j < 4; ++j)
#pragma unroll
      for (int r = 0; r < 8; ++r) {
        int m = m0 + wm + i * 16 + half * 8 + r;
        int n = n0 + wn + j * 16 + nn;
        out[(size_t)m * EMB + n] = acc[i][j][r];
      }
}

// ---------------------------------------------------------------------------
extern "C" void kernel_launch(void* const* d_in, const int* in_sizes, int n_in,
                              void* d_out, int out_size, void* d_ws, size_t ws_size,
                              hipStream_t stream) {
  (void)in_sizes; (void)n_in; (void)out_size; (void)ws_size;
  const float* inputs_q  = (const float*)d_in[0];
  const float* inputs_kv = (const float*)d_in[1];
  // d_in[2] = mask (bool, causal tril) — applied analytically in-kernel.
  const float* Wq = (const float*)d_in[3];
  const float* Wk = (const float*)d_in[4];
  const float* Wv = (const float*)d_in[5];
  const float* Wo = (const float*)d_in[6];
  const float* rel_bias = (const float*)d_in[7];

  const size_t per = (size_t)4 * NH * S_LEN * HD;  // 8 Mi bf16 elements
  bf16* q_ws    = (bf16*)d_ws;
  bf16* k_ws    = q_ws + per;
  bf16* v_ws    = k_ws + per;
  bf16* attn_ws = v_ws + per;

  qkv_gemm_kernel<<<dim3(8192 / TM, EMB / TN, 3), 256, 0, stream>>>(
      inputs_q, inputs_kv, Wq, Wk, Wv, q_ws, k_ws, v_ws);
  flash_attn_kernel<<<dim3(S_LEN / QT, 4 * NH), 128, 0, stream>>>(
      q_ws, k_ws, v_ws, rel_bias, attn_ws);
  out_gemm_kernel<<<dim3(8192 / TM, EMB / TN), 256, 0, stream>>>(
      attn_ws, Wo, (float*)d_out);
}
